// TFAttention_28621662060547
// MI455X (gfx1250) — matrix operand order repaired and checked by reading, not compiled, not run on hardware
//
#include <hip/hip_runtime.h>
#include <hip/hip_bf16.h>

typedef __bf16 bf16_t;
typedef __attribute__((ext_vector_type(16))) __bf16 v16bf;
typedef __attribute__((ext_vector_type(4)))  __bf16 v4bf;
typedef __attribute__((ext_vector_type(8)))  float  v8f;
typedef __attribute__((ext_vector_type(4)))  float  f32x4;

#define NXC   1024
#define SEQ   2048
#define BATCH 4
#define NHEAD 16
#define HDIM  64

// ---------------------------------------------------------------------------
// WMMA helper: D = A(16x32 bf16) * B(32x16 bf16) + C(16x16 f32)
// ---------------------------------------------------------------------------
__device__ __forceinline__ v8f wmma_bf16(v16bf a, v16bf b, v8f c) {
    return __builtin_amdgcn_wmma_f32_16x16x32_bf16(
        false, a, false, b, (short)0, c, false, false);
}

// A fragment (16x32, MxK), A row-major with leading dim `ld`:
// lane 0-15: row M=lane,  K = {0..7, 16..23}
// lane 16-31: row M=lane-16, K = {8..15, 24..31}
__device__ __forceinline__ v16bf load_frag_a(const bf16_t* base, int ld,
                                             int row0, int k0, int lane) {
    int lm = lane & 15, hi = lane >> 4;
    const bf16_t* p = base + (long)(row0 + lm) * ld + k0 + hi * 8;
    v16bf a;
    f32x4* d = (f32x4*)&a;
    d[0] = *(const f32x4*)(p);        // K = k0+off    .. +7
    d[1] = *(const f32x4*)(p + 16);   // K = k0+16+off .. +23
    return a;
}

// B fragment (32x16, KxN) sourced from an (N,K) row-major array (i.e. B^T):
// lane 0-15: N=lane, K = {0..7, 16..23} ; lane 16-31: N=lane-16, K = {8..15, 24..31}
__device__ __forceinline__ v16bf load_frag_b(const bf16_t* baseT, int ld,
                                             int n0, int k0, int lane) {
    int ln = lane & 15, kh = lane >> 4;
    const bf16_t* p = baseT + (long)(n0 + ln) * ld + k0 + kh * 8;
    v16bf b;
    f32x4* d = (f32x4*)&b;
    d[0] = *(const f32x4*)(p);        // K = k0+kh*8     .. +7
    d[1] = *(const f32x4*)(p + 16);   // K = k0+16+kh*8  .. +7
    return b;
}

// ---------------------------------------------------------------------------
// Preprocessing: f32 -> bf16 convert (vectorized) and tiled transpose
// ---------------------------------------------------------------------------
__global__ __launch_bounds__(256) void k_convert_bf16(
        const float* __restrict__ X, bf16_t* __restrict__ Y, long n4) {
    long i = (long)blockIdx.x * blockDim.x + threadIdx.x;
    if (i >= n4) return;
    f32x4 v = *(const f32x4*)(X + i * 4);
    v4bf o = { (bf16_t)v.x, (bf16_t)v.y, (bf16_t)v.z, (bf16_t)v.w };
    *(v4bf*)(Y + i * 4) = o;
}

// W (K,N) row-major f32 -> Wt (N,K) row-major bf16, 32x32 LDS tiles.
// grid = (N/32, K/32), 256 threads.
__global__ __launch_bounds__(256) void k_transpose_bf16(
        const float* __restrict__ W, bf16_t* __restrict__ Wt, int K, int N) {
    __shared__ bf16_t tile[32][33];
    const int nb = blockIdx.x * 32, kb = blockIdx.y * 32;
    const int tx = threadIdx.x & 31, ty = threadIdx.x >> 5;   // ty = 0..7
#pragma unroll
    for (int i = 0; i < 4; ++i) {
        int k = kb + ty + 8 * i;
        tile[ty + 8 * i][tx] = (bf16_t)W[(long)k * N + nb + tx];
    }
    __syncthreads();
#pragma unroll
    for (int i = 0; i < 4; ++i) {
        int n = nb + ty + 8 * i;
        Wt[(long)n * K + kb + tx] = tile[tx][ty + 8 * i];
    }
}

// ---------------------------------------------------------------------------
// QKV GEMM: (8192 x 1024) @ (1024 x 3072) + bias, scattered into
// Q,K : (B,H,S,D) bf16    Vt : (B,H,D,S) bf16
// block = 8 waves (2 M x 4 N), wave tile 64x64, block tile 128x256
// ---------------------------------------------------------------------------
__global__ __launch_bounds__(256) void k_gemm_qkv(
        const bf16_t* __restrict__ A, const bf16_t* __restrict__ Bt,
        const float* __restrict__ bias,
        bf16_t* __restrict__ Q, bf16_t* __restrict__ Kk,
        bf16_t* __restrict__ Vt) {
    const int lane = threadIdx.x & 31, wid = threadIdx.x >> 5;
    const int m0 = blockIdx.x * 128 + (wid & 1) * 64;
    const int n0 = blockIdx.y * 256 + (wid >> 1) * 64;

    v8f acc[4][4] = {};
    for (int k0 = 0; k0 < NXC; k0 += 32) {
        v16bf af[4], bf[4];
#pragma unroll
        for (int mt = 0; mt < 4; ++mt) {
            af[mt] = load_frag_a(A, NXC, m0 + mt * 16, k0, lane);
            __builtin_prefetch(A + (long)(m0 + mt * 16 + (lane & 15)) * NXC + k0 + 256, 0, 1);
        }
#pragma unroll
        for (int nt = 0; nt < 4; ++nt) {
            bf[nt] = load_frag_b(Bt, NXC, n0 + nt * 16, k0, lane);
            __builtin_prefetch(Bt + (long)(n0 + nt * 16 + (lane & 15)) * NXC + k0 + 256, 0, 1);
        }
#pragma unroll
        for (int mt = 0; mt < 4; ++mt)
#pragma unroll
            for (int nt = 0; nt < 4; ++nt)
                acc[mt][nt] = wmma_bf16(af[mt], bf[nt], acc[mt][nt]);
    }

    const int ln = lane & 15, hi = lane >> 4;
#pragma unroll
    for (int mt = 0; mt < 4; ++mt)
#pragma unroll
        for (int nt = 0; nt < 4; ++nt) {
            int n = n0 + nt * 16 + ln;
            float bv = bias[n];
            int sec = n >> 10;          // 0=Q 1=K 2=V
            int c = n & (NXC - 1);
            int h = c >> 6, d = c & 63;
#pragma unroll
            for (int r = 0; r < 8; ++r) {
                int m = m0 + mt * 16 + r + 8 * hi;   // global token row
                int b = m >> 11, s = m & (SEQ - 1);
                bf16_t v = (bf16_t)(acc[mt][nt][r] + bv);
                long bh = (long)b * NHEAD + h;
                if (sec == 0)      Q [(bh * SEQ + s) * HDIM + d] = v;
                else if (sec == 1) Kk[(bh * SEQ + s) * HDIM + d] = v;
                else               Vt[(bh * HDIM + d) * SEQ + s] = v;
            }
        }
}

// ---------------------------------------------------------------------------
// Flash attention: one wave = 16 query rows, loop keys in steps of 32.
// Scores scaled by 1/8, additive mask, online softmax, P@V accumulate.
// Output merged-heads bf16 (B*S, NX).
// ---------------------------------------------------------------------------
__global__ __launch_bounds__(128) void k_attention(
        const bf16_t* __restrict__ Q, const bf16_t* __restrict__ Kk,
        const bf16_t* __restrict__ Vt, const float* __restrict__ mask,
        bf16_t* __restrict__ Am) {
    __shared__ bf16_t Pbuf[4][16 * 32];

    const int lane = threadIdx.x & 31, wid = threadIdx.x >> 5;
    const int bh = blockIdx.x;               // b*NHEAD + h
    const int b = bh >> 4, h = bh & 15;
    const int q0 = (blockIdx.y * 4 + wid) * 16;
    const int ln = lane & 15, hi = lane >> 4;

    const bf16_t* Qh = Q  + (long)bh * SEQ * HDIM;
    const bf16_t* Kh = Kk + (long)bh * SEQ * HDIM;
    const bf16_t* Vh = Vt + (long)bh * HDIM * SEQ;
    const float*  mb = mask + (long)b * SEQ;

    // Q fragments for this wave's 16 rows (K-dim = HDIM = 64 -> 2 frags)
    v16bf qa0 = load_frag_a(Qh, HDIM, q0, 0,  lane);
    v16bf qa1 = load_frag_a(Qh, HDIM, q0, 32, lane);

    v8f O[4] = {};
    float mrun[8], lrun[8];
#pragma unroll
    for (int r = 0; r < 8; ++r) { mrun[r] = -1e30f; lrun[r] = 0.f; }

    bf16_t* P = Pbuf[wid];

    for (int j = 0; j < SEQ; j += 32) {
        v8f s0 = {}, s1 = {};
        v16bf kb;
        kb = load_frag_b(Kh, HDIM, j,      0,  lane); s0 = wmma_bf16(qa0, kb, s0);
        kb = load_frag_b(Kh, HDIM, j,      32, lane); s0 = wmma_bf16(qa1, kb, s0);
        kb = load_frag_b(Kh, HDIM, j + 16, 0,  lane); s1 = wmma_bf16(qa0, kb, s1);
        kb = load_frag_b(Kh, HDIM, j + 16, 32, lane); s1 = wmma_bf16(qa1, kb, s1);

        float mv0 = mb[j + ln];
        float mv1 = mb[j + 16 + ln];
        float alpha[8];
#pragma unroll
        for (int r = 0; r < 8; ++r) {
            float t0 = s0[r] * 0.125f + mv0;   // 1/sqrt(64)
            float t1 = s1[r] * 0.125f + mv1;
            // row max over the 16 lanes of this half-wave
            float rm = fmaxf(t0, t1);
            rm = fmaxf(rm, __shfl_xor(rm, 1, 32));
            rm = fmaxf(rm, __shfl_xor(rm, 2, 32));
            rm = fmaxf(rm, __shfl_xor(rm, 4, 32));
            rm = fmaxf(rm, __shfl_xor(rm, 8, 32));
            float mnew = fmaxf(mrun[r], rm);
            float al = __expf(mrun[r] - mnew);
            float p0 = __expf(t0 - mnew);
            float p1 = __expf(t1 - mnew);
            float rs = p0 + p1;
            rs += __shfl_xor(rs, 1, 32);
            rs += __shfl_xor(rs, 2, 32);
            rs += __shfl_xor(rs, 4, 32);
            rs += __shfl_xor(rs, 8, 32);
            lrun[r] = lrun[r] * al + rs;
            mrun[r] = mnew;
            alpha[r] = al;
            int row = r + 8 * hi;              // C-layout row of this VGPR
            P[row * 32 + ln]      = (bf16_t)p0;
            P[row * 32 + 16 + ln] = (bf16_t)p1;
        }
#pragma unroll
        for (int t = 0; t < 4; ++t)
#pragma unroll
            for (int r = 0; r < 8; ++r) O[t][r] *= alpha[r];

        // reload P in A-fragment layout (row-major 16x32 in LDS, stride 32)
        v16bf pa;
        f32x4* pd = (f32x4*)&pa;
        pd[0] = *(const f32x4*)(P + ln * 32 + hi * 8);
        pd[1] = *(const f32x4*)(P + ln * 32 + hi * 8 + 16);

#pragma unroll
        for (int t = 0; t < 4; ++t) {
            v16bf vb = load_frag_b(Vh, SEQ, t * 16, j, lane); // N=d, K=keys
            O[t] = wmma_bf16(pa, vb, O[t]);
        }
    }

    // normalize and write merged heads (B*S, NX)
#pragma unroll
    for (int t = 0; t < 4; ++t)
#pragma unroll
        for (int r = 0; r < 8; ++r) {
            float v = O[t][r] / lrun[r];
            int srow = q0 + r + 8 * hi;
            long m = (long)b * SEQ + srow;
            Am[m * NXC + h * HDIM + t * 16 + ln] = (bf16_t)v;
        }
}

// ---------------------------------------------------------------------------
// Output projection: (8192 x 1024) @ (1024 x 1024) + bias -> f32 out
// block = 8 waves (2 M x 4 N), wave tile 64x64
// ---------------------------------------------------------------------------
__global__ __launch_bounds__(256) void k_gemm_proj(
        const bf16_t* __restrict__ A, const bf16_t* __restrict__ Bt,
        const float* __restrict__ bias, float* __restrict__ out) {
    const int lane = threadIdx.x & 31, wid = threadIdx.x >> 5;
    const int m0 = blockIdx.x * 128 + (wid & 1) * 64;
    const int n0 = blockIdx.y * 256 + (wid >> 1) * 64;

    v8f acc[4][4] = {};
    for (int k0 = 0; k0 < NXC; k0 += 32) {
        v16bf af[4], bf[4];
#pragma unroll
        for (int mt = 0; mt < 4; ++mt) {
            af[mt] = load_frag_a(A, NXC, m0 + mt * 16, k0, lane);
            __builtin_prefetch(A + (long)(m0 + mt * 16 + (lane & 15)) * NXC + k0 + 256, 0, 1);
        }
#pragma unroll
        for (int nt = 0; nt < 4; ++nt) {
            bf[nt] = load_frag_b(Bt, NXC, n0 + nt * 16, k0, lane);
            __builtin_prefetch(Bt + (long)(n0 + nt * 16 + (lane & 15)) * NXC + k0 + 256, 0, 1);
        }
#pragma unroll
        for (int mt = 0; mt < 4; ++mt)
#pragma unroll
            for (int nt = 0; nt < 4; ++nt)
                acc[mt][nt] = wmma_bf16(af[mt], bf[nt], acc[mt][nt]);
    }

    const int ln = lane & 15, hi = lane >> 4;
#pragma unroll
    for (int mt = 0; mt < 4; ++mt)
#pragma unroll
        for (int nt = 0; nt < 4; ++nt) {
            int n = n0 + nt * 16 + ln;
            float bv = bias[n];
#pragma unroll
            for (int r = 0; r < 8; ++r) {
                int m = m0 + mt * 16 + r + 8 * hi;
                out[(long)m * NXC + n] = acc[mt][nt][r] + bv;
            }
        }
}

// ---------------------------------------------------------------------------
extern "C" void kernel_launch(void* const* d_in, const int* in_sizes, int n_in,
                              void* d_out, int out_size, void* d_ws, size_t ws_size,
                              hipStream_t stream) {
    const float* x        = (const float*)d_in[0];   // (4,2048,1024)
    const float* amask    = (const float*)d_in[1];   // (4,1,1,2048)
    const float* c_attn_w = (const float*)d_in[2];   // (1024,3072)
    const float* c_attn_b = (const float*)d_in[3];   // (3072)
    const float* c_proj_w = (const float*)d_in[4];   // (1024,1024)
    const float* c_proj_b = (const float*)d_in[5];   // (1024)
    float* out = (float*)d_out;

    const long M = (long)BATCH * SEQ;                // 8192
    char* ws = (char*)d_ws;
    size_t off = 0;
    bf16_t* xb  = (bf16_t*)(ws + off); off += (size_t)M * NXC * 2;        // 16 MB (reused as Am)
    bf16_t* wat = (bf16_t*)(ws + off); off += (size_t)3 * NXC * NXC * 2;  //  6 MB
    bf16_t* wpt = (bf16_t*)(ws + off); off += (size_t)NXC * NXC * 2;      //  2 MB
    bf16_t* Qb  = (bf16_t*)(ws + off); off += (size_t)M * NXC * 2;        // 16 MB
    bf16_t* Kb  = (bf16_t*)(ws + off); off += (size_t)M * NXC * 2;        // 16 MB
    bf16_t* Vtb = (bf16_t*)(ws + off); off += (size_t)M * NXC * 2;        // 16 MB
    bf16_t* Am  = xb;  // x no longer needed after QKV GEMM

    // 1) convert x to bf16 (4 elems / thread)
    {
        long n4 = (M * NXC) / 4;
        k_convert_bf16<<<dim3((n4 + 255) / 256), dim3(256), 0, stream>>>(x, xb, n4);
    }
    // 2) transpose + convert weights (32x32 LDS tiles)
    k_transpose_bf16<<<dim3((3 * NXC) / 32, NXC / 32), dim3(256), 0, stream>>>(
        c_attn_w, wat, NXC, 3 * NXC);
    k_transpose_bf16<<<dim3(NXC / 32, NXC / 32), dim3(256), 0, stream>>>(
        c_proj_w, wpt, NXC, NXC);
    // 3) QKV GEMM with head-split scatter
    k_gemm_qkv<<<dim3(M / 128, (3 * NXC) / 256), dim3(256), 0, stream>>>(
        xb, wat, c_attn_b, Qb, Kb, Vtb);
    // 4) attention (flash, online softmax)
    k_attention<<<dim3(BATCH * NHEAD, SEQ / 64), dim3(128), 0, stream>>>(
        Qb, Kb, Vtb, amask, Am);
    // 5) output projection
    k_gemm_proj<<<dim3(M / 128, NXC / 256), dim3(256), 0, stream>>>(
        Am, wpt, c_proj_b, out);
}
